// APQBAttention_16870631539232
// MI455X (gfx1250) — matrix operand order repaired and checked by reading, not compiled, not run on hardware
//
#include <hip/hip_runtime.h>
#include <hip/hip_bf16.h>

// ---------------------------------------------------------------------------
// APQB attention for MI455X (gfx1250, wave32, WMMA).
//
// Pipeline:
//   (1) tmean_kernel          : T_mean = mean_h |sin(2*sigmoid(theta)*pi/2)|
//   (2) gemm_wmma<f32->QKV>   : q,k = x@W^T+b -> bf16 [B,H,T,D]
//       gemm_wmma<f32->VT>    : v   = x@W^T+b -> bf16 [B,H,D,T] (transposed!)
//   (3) attn_kernel           : flash-style online softmax + post-softmax
//                               dropout (noise > T_mean); K/V^T tiles staged
//                               block-wide via GLOBAL_LOAD_ASYNC_TO_LDS with
//                               double buffering + partial s_wait_asynccnt
//   (4) gemm_wmma<bf16->f32>  : out = y @ Wo^T + bo -> f32 d_out
//
// The per-head additive bias (corr_w/theta term) is constant along the
// softmax axis -> cancels exactly in softmax -> intentionally omitted.
//
// GEMM: 128x128 block tile, 8 waves each 32x64 (8 WMMA/k-step), register
// prefetch of the next K-tile so global loads overlap WMMA compute.
// All WMMA fragment accesses are contiguous 16B runs -> ds_*_b128 traffic.
//
// Workspace requirement: 4 * 8MB bf16 buffers + 4 bytes (~32MB).
// ---------------------------------------------------------------------------

typedef __attribute__((ext_vector_type(16))) __bf16 v16bf;
typedef __attribute__((ext_vector_type(8)))  __bf16 v8bf;
typedef __attribute__((ext_vector_type(8)))  float  v8f;

#define WMMA_BF16(a, b, c) \
  __builtin_amdgcn_wmma_f32_16x16x32_bf16(false, (a), false, (b), (short)0, (c), false, false)

namespace {

constexpr int kE = 1024;   // embed dim
constexpr int kH = 16;     // heads
constexpr int kD = 64;     // head dim
constexpr int kB = 4;      // batch
constexpr int kT = 1024;   // sequence
constexpr int kM = kB * kT;

enum OutMode { OUT_F32 = 0, OUT_QKV = 1, OUT_VT = 2 };

__device__ __forceinline__ v16bf cat8(v8bf lo, v8bf hi) {
  return __builtin_shufflevector(lo, hi, 0, 1, 2, 3, 4, 5, 6, 7, 8, 9, 10, 11, 12, 13, 14, 15);
}

// LDS byte offset of a generic pointer into shared memory (for async-to-LDS).
typedef __attribute__((address_space(3))) const void* as3_cptr;
__device__ __forceinline__ unsigned lds_off(const void* p) {
  return (unsigned)(unsigned long long)(as3_cptr)p;
}

// --- WMMA fragment loaders (contiguous k order, same in A and B) -------
// A 16x32 bf16: lane&15 -> M, element e => K = (lane>>4)*16 + e (matches B).
//               One contiguous 32B run (2x16B loads).
__device__ __forceinline__ v16bf frag_a_lds(const __bf16* base, int stride, int lane) {
  const __bf16* row = base + (size_t)(lane & 15) * stride + (lane >> 4) * 16;
  return cat8(*(const v8bf*)row, *(const v8bf*)(row + 8));
}
// B 32x16 bf16: lane&15 -> N, lanes0..15 hold K=0..15, lanes16..31 K=16..31,
// element e => K = (lane>>4)*16 + e.  One contiguous 32B run (2x16B loads).
__device__ __forceinline__ v16bf frag_b_lds(const __bf16* base, int stride, int lane) {
  const __bf16* row = base + (size_t)(lane & 15) * stride + (lane >> 4) * 16;
  return cat8(*(const v8bf*)row, *(const v8bf*)(row + 8));
}

// --- (1) scalar dropout threshold -------------------------------------------
__global__ void tmean_kernel(const float* __restrict__ theta, float* __restrict__ tm) {
  const int l = threadIdx.x;
  float v = 0.0f;
  if (l < kH) {
    const float sig = 1.0f / (1.0f + __expf(-theta[l]));
    const float th  = sig * 1.5707963267948966f;  // pi/2
    v = fabsf(__sinf(2.0f * th));
  }
#pragma unroll
  for (int off = 1; off < 16; off <<= 1) v += __shfl_xor(v, off, 32);
  if (l == 0) tm[0] = v * (1.0f / 16.0f);
}

// --- (2)/(4) GEMM: out[m,n] = sum_k in[m,k] * W[n,k] + bias[n] --------------
// Block tile 128(M) x 128(N), K-step 32, 8 waves (4x2), wave = 32x64 (2x4 WMMA).
// Register-prefetch of the next K-tile overlaps global loads with WMMAs.
template <bool IN_BF16, int OUT_MODE>
__global__ __launch_bounds__(256) void gemm_wmma_kernel(
    const void* __restrict__ inp, const float* __restrict__ W,
    const float* __restrict__ bias, void* __restrict__ outp) {
  constexpr int K = kE;
  constexpr int LDA = 40;  // +8 halves pad (rows stay 16B aligned: 80B stride)
  __shared__ __bf16 As[128][LDA];
  __shared__ __bf16 Bs[128][LDA];

  const int tid  = threadIdx.x;
  const int lane = tid & 31;
  const int wave = tid >> 5;
  const int wm   = (wave >> 1) * 32;
  const int wn   = (wave & 1) * 64;
  const int m0   = blockIdx.x * 128;
  const int n0   = blockIdx.y * 128;

  v8f acc[2][4];
#pragma unroll
  for (int i = 0; i < 2; ++i)
#pragma unroll
    for (int j = 0; j < 4; ++j) acc[i][j] = (v8f){0, 0, 0, 0, 0, 0, 0, 0};

  const int arow = tid >> 1, aseg = (tid & 1) * 16;  // A: 128x32, 16 halves/thread
  const int brow = tid >> 1, bseg = (tid & 1) * 16;  // B: 128x32, 16 halves/thread

  // ---- register prefetch state for the next K-tile -------------------------
  float  raf[16], rbf[16];  // f32 input path
  v8bf   ra0, ra1;          // bf16 input path (A only; W is always f32)

  auto load_tile = [&](int kk) {
    if constexpr (IN_BF16) {
      const __bf16* srcA = (const __bf16*)inp + (size_t)(m0 + arow) * K + kk + aseg;
      ra0 = *(const v8bf*)srcA;
      ra1 = *(const v8bf*)(srcA + 8);
    } else {
      const float* srcA = (const float*)inp + (size_t)(m0 + arow) * K + kk + aseg;
#pragma unroll
      for (int e = 0; e < 16; ++e) raf[e] = srcA[e];
    }
    const float* srcB = W + (size_t)(n0 + brow) * K + kk + bseg;
#pragma unroll
    for (int e = 0; e < 16; ++e) rbf[e] = srcB[e];
  };

  auto store_tile = [&]() {
    if constexpr (IN_BF16) {
      *(v8bf*)&As[arow][aseg]     = ra0;
      *(v8bf*)&As[arow][aseg + 8] = ra1;
    } else {
      v8bf t0, t1;
#pragma unroll
      for (int e = 0; e < 8; ++e) { t0[e] = (__bf16)raf[e]; t1[e] = (__bf16)raf[e + 8]; }
      *(v8bf*)&As[arow][aseg]     = t0;
      *(v8bf*)&As[arow][aseg + 8] = t1;
    }
    v8bf u0, u1;
#pragma unroll
    for (int e = 0; e < 8; ++e) { u0[e] = (__bf16)rbf[e]; u1[e] = (__bf16)rbf[e + 8]; }
    *(v8bf*)&Bs[brow][bseg]     = u0;
    *(v8bf*)&Bs[brow][bseg + 8] = u1;
  };

  load_tile(0);
  for (int kk = 0; kk < K; kk += 32) {
    store_tile();
    __syncthreads();
    if (kk + 32 < K) load_tile(kk + 32);  // globals fly while WMMAs run

    const v16bf a0 = frag_a_lds(&As[wm][0], LDA, lane);
    const v16bf a1 = frag_a_lds(&As[wm + 16][0], LDA, lane);
#pragma unroll
    for (int j = 0; j < 4; ++j) {
      const v16bf bj = frag_b_lds(&Bs[wn + j * 16][0], LDA, lane);
      acc[0][j] = WMMA_BF16(a0, bj, acc[0][j]);
      acc[1][j] = WMMA_BF16(a1, bj, acc[1][j]);
    }
    __syncthreads();
  }

  // C layout: lane<16 -> N=lane, M=r; lane>=16 -> N=lane-16, M=8+r.
  const int mrow = 8 * (lane >> 4);
  const int ncol = lane & 15;
#pragma unroll
  for (int i = 0; i < 2; ++i)
#pragma unroll
    for (int j = 0; j < 4; ++j) {
      const int gn = n0 + wn + j * 16 + ncol;
      const float bb = bias[gn];
#pragma unroll
      for (int r = 0; r < 8; ++r) {
        const int gm = m0 + wm + i * 16 + mrow + r;
        const float val = acc[i][j][r] + bb;
        if constexpr (OUT_MODE == OUT_QKV) {
          const int b = gm >> 10, t = gm & (kT - 1);
          const int h = gn >> 6,  d = gn & (kD - 1);
          ((__bf16*)outp)[((size_t)(b * kH + h) * kT + t) * kD + d] = (__bf16)val;
        } else if constexpr (OUT_MODE == OUT_VT) {
          const int b = gm >> 10, t = gm & (kT - 1);
          const int h = gn >> 6,  d = gn & (kD - 1);
          ((__bf16*)outp)[((size_t)(b * kH + h) * kD + d) * kT + t] = (__bf16)val;
        } else {
          ((float*)outp)[(size_t)gm * kE + gn] = val;
        }
      }
    }
}

// --- (3) flash-style attention ----------------------------------------------
// grid = (B*H, T/128); 8 waves/block; each wave owns 16 query rows.
// K/V^T tiles async-staged block-wide into double-buffered LDS; chunk i+1's
// copies are issued before chunk i's compute, consumed after a PARTIAL
// s_wait_asynccnt (ASYNCcnt completes in order per CDNA5 ISA).
__global__ __launch_bounds__(256) void attn_kernel(
    const __bf16* __restrict__ q, const __bf16* __restrict__ k,
    const __bf16* __restrict__ vt, const float* __restrict__ noise,
    const float* __restrict__ tm_ptr, __bf16* __restrict__ y) {
  const float scale = 0.125f;  // 1/sqrt(64)
  const int bh   = blockIdx.x;  // b*H + h
  const int tid  = threadIdx.x;
  const int wave = tid >> 5;
  const int lane = tid & 31;
  const int t0   = blockIdx.y * 128 + wave * 16;
  const int b    = bh >> 4, h = bh & 15;

  const __bf16* qh  = q  + (size_t)bh * kT * kD;
  const __bf16* kh  = k  + (size_t)bh * kT * kD;
  const __bf16* vth = vt + (size_t)bh * kD * kT;
  const float*  nz  = noise + (size_t)bh * kT * kT + (size_t)t0 * kT;
  const float Tm = *tm_ptr;

  __shared__ __bf16 Ks[2][32][72];      // K tile  [s_local][d]  (+pad, 144B rows)
  __shared__ __bf16 Vts[2][64][40];     // V^T tile [d][s_local] (+pad, 80B rows)
  __shared__ __bf16 Pbuf[8][16][40];    // per-wave P tile (A-layout source)

  // staging coordinates: one 16B async transfer per lane per tile
  const int krow = tid >> 3, kseg = (tid & 7) * 8;  // 32 x 64
  const int vrow = tid >> 2, vseg = (tid & 3) * 8;  // 64 x 32

  auto issue_chunk = [&](int s0, int buf) {
    const unsigned kdst = lds_off(&Ks[buf][krow][kseg]);
    const __bf16* kg = kh + (size_t)(s0 + krow) * kD + kseg;
    asm volatile("global_load_async_to_lds_b128 %0, %1, off"
                 :: "v"(kdst), "v"(kg) : "memory");
    const unsigned vdst = lds_off(&Vts[buf][vrow][vseg]);
    const __bf16* vg = vth + (size_t)vrow * kT + s0 + vseg;
    asm volatile("global_load_async_to_lds_b128 %0, %1, off"
                 :: "v"(vdst), "v"(vg) : "memory");
  };

  // q A-fragments, K=64 -> two 16x32 frags (contiguous 16B runs, loaded once).
  v16bf aq[2];
  {
    const __bf16* row = qh + (size_t)(t0 + (lane & 15)) * kD + (lane >> 4) * 16;
#pragma unroll
    for (int kt = 0; kt < 2; ++kt)
      aq[kt] = cat8(*(const v8bf*)(row + kt * 32), *(const v8bf*)(row + kt * 32 + 8));
  }

  v8f o[4];
#pragma unroll
  for (int j = 0; j < 4; ++j) o[j] = (v8f){0, 0, 0, 0, 0, 0, 0, 0};
  float rmax[8], rsum[8];
#pragma unroll
  for (int r = 0; r < 8; ++r) { rmax[r] = -__builtin_inff(); rsum[r] = 0.0f; }

  issue_chunk(0, 0);
  for (int s0 = 0; s0 < kT; s0 += 32) {
    const int buf = (s0 >> 5) & 1;
    if (s0 + 32 < kT) {
      issue_chunk(s0 + 32, buf ^ 1);  // overlap next chunk's DMA with compute
      asm volatile("s_wait_asynccnt 0x2" ::: "memory");  // chunk s0 landed
    } else {
      asm volatile("s_wait_asynccnt 0x0" ::: "memory");
    }
    __syncthreads();

    // ---- scores: two 16x16 tiles over K=64 --------------------------------
    v8f sc[2];
#pragma unroll
    for (int j = 0; j < 2; ++j) {
      sc[j] = (v8f){0, 0, 0, 0, 0, 0, 0, 0};
      const __bf16* srow = &Ks[buf][16 * j + (lane & 15)][(lane >> 4) * 16];
#pragma unroll
      for (int kt = 0; kt < 2; ++kt) {
        const v16bf bk = cat8(*(const v8bf*)(srow + kt * 32), *(const v8bf*)(srow + kt * 32 + 8));
        sc[j] = WMMA_BF16(aq[kt], bk, sc[j]);
      }
#pragma unroll
      for (int r = 0; r < 8; ++r) sc[j][r] *= scale;
    }
    // Per-head softmax bias omitted: constant along s -> cancels in softmax.

    // ---- online softmax (row stats via xor-shuffles in 16-lane halves) ----
    float cm[8];
#pragma unroll
    for (int r = 0; r < 8; ++r) cm[r] = fmaxf(sc[0][r], sc[1][r]);
#pragma unroll
    for (int off = 1; off < 16; off <<= 1)
#pragma unroll
      for (int r = 0; r < 8; ++r) cm[r] = fmaxf(cm[r], __shfl_xor(cm[r], off, 32));

    float corr[8];
#pragma unroll
    for (int r = 0; r < 8; ++r) {
      const float nm = fmaxf(rmax[r], cm[r]);
      corr[r] = __expf(rmax[r] - nm);
      rmax[r] = nm;
    }

    float psum[8];
#pragma unroll
    for (int r = 0; r < 8; ++r) psum[r] = 0.0f;
#pragma unroll
    for (int j = 0; j < 2; ++j) {
      const int s = s0 + 16 * j + (lane & 15);
#pragma unroll
      for (int r = 0; r < 8; ++r) {
        const int trow = 8 * (lane >> 4) + r;
        const float p = __expf(sc[j][r] - rmax[r]);
        psum[r] += p;  // Z accumulates UNMASKED p
        const float nv = nz[(size_t)trow * kT + s];
        Pbuf[wave][trow][16 * j + (lane & 15)] = (__bf16)((nv > Tm) ? p : 0.0f);
      }
    }
#pragma unroll
    for (int off = 1; off < 16; off <<= 1)
#pragma unroll
      for (int r = 0; r < 8; ++r) psum[r] += __shfl_xor(psum[r], off, 32);
#pragma unroll
    for (int r = 0; r < 8; ++r) rsum[r] = rsum[r] * corr[r] + psum[r];
#pragma unroll
    for (int j = 0; j < 4; ++j)
#pragma unroll
      for (int r = 0; r < 8; ++r) o[j][r] *= corr[r];

    // same-wave LDS RAW: DS pipe is in-order; wait before fragment read.
    asm volatile("s_wait_dscnt 0x0" ::: "memory");

    // P -> A-fragment (two contiguous 16B runs)
    const __bf16* prow = &Pbuf[wave][lane & 15][(lane >> 4) * 16];
    const v16bf pf = cat8(*(const v8bf*)prow, *(const v8bf*)(prow + 8));

    // o += P @ V : four 16-wide d tiles, B-frags contiguous from V^T tile
#pragma unroll
    for (int jn = 0; jn < 4; ++jn) {
      const __bf16* vrow_p = &Vts[buf][jn * 16 + (lane & 15)][(lane >> 4) * 16];
      const v16bf bv = cat8(*(const v8bf*)vrow_p, *(const v8bf*)(vrow_p + 8));
      o[jn] = WMMA_BF16(pf, bv, o[jn]);
    }
    __syncthreads();  // seal this buffer before chunk s0+64 overwrites it
  }

  // out = o / (Z * (1 - Tm + 1e-8)); store [B,T,E] bf16 for the final GEMM.
  const float inv = 1.0f / (1.0f - Tm + 1e-8f);
#pragma unroll
  for (int jn = 0; jn < 4; ++jn)
#pragma unroll
    for (int r = 0; r < 8; ++r) {
      const int t = t0 + 8 * (lane >> 4) + r;
      const int d = jn * 16 + (lane & 15);
      const float val = o[jn][r] / rsum[r] * inv;
      y[((size_t)(b * kT + t)) * kE + h * kD + d] = (__bf16)val;
    }
}

}  // namespace

extern "C" void kernel_launch(void* const* d_in, const int* in_sizes, int n_in,
                              void* d_out, int out_size, void* d_ws, size_t ws_size,
                              hipStream_t stream) {
  (void)in_sizes; (void)n_in; (void)out_size; (void)ws_size;
  const float* x     = (const float*)d_in[0];
  const float* noise = (const float*)d_in[1];
  const float* Wq    = (const float*)d_in[2];
  const float* bq    = (const float*)d_in[3];
  const float* Wk    = (const float*)d_in[4];
  const float* bk    = (const float*)d_in[5];
  const float* Wv    = (const float*)d_in[6];
  const float* bv    = (const float*)d_in[7];
  const float* Wo    = (const float*)d_in[8];
  const float* bo    = (const float*)d_in[9];
  const float* theta = (const float*)d_in[10];
  // d_in[11] (corr_w) only feeds the softmax-invariant bias -> unused.

  constexpr size_t kBuf = (size_t)kM * kE;
  __bf16* qb = (__bf16*)d_ws;
  __bf16* kb = qb + kBuf;
  __bf16* vb = kb + kBuf;   // V^T layout [B,H,D,T]
  __bf16* yb = vb + kBuf;
  float*  tm = (float*)(yb + kBuf);

  tmean_kernel<<<1, 32, 0, stream>>>(theta, tm);

  const dim3 gg(kM / 128, kE / 128);  // (32, 8)
  gemm_wmma_kernel<false, OUT_QKV><<<gg, 256, 0, stream>>>(x, Wq, bq, qb);
  gemm_wmma_kernel<false, OUT_QKV><<<gg, 256, 0, stream>>>(x, Wk, bk, kb);
  gemm_wmma_kernel<false, OUT_VT ><<<gg, 256, 0, stream>>>(x, Wv, bv, vb);

  const dim3 ga(kB * kH, kT / 128);  // (64, 8)
  attn_kernel<<<ga, 256, 0, stream>>>(qb, kb, vb, noise, tm, yb);

  gemm_wmma_kernel<true, OUT_F32><<<gg, 256, 0, stream>>>(yb, Wo, bo, (float*)d_out);
}
